// MultiheadAttention_86517821216099
// MI455X (gfx1250) — matrix operand order repaired and checked by reading, not compiled, not run on hardware
//
#include <hip/hip_runtime.h>
#include <hip/hip_bf16.h>

typedef __attribute__((ext_vector_type(16))) __bf16 v16bf;
typedef __attribute__((ext_vector_type(4)))  __bf16 v4bf;
typedef __attribute__((ext_vector_type(2)))  __bf16 v2bf;
typedef __attribute__((ext_vector_type(8)))  float  v8f;
typedef __attribute__((ext_vector_type(4)))  float  v4f;
typedef __attribute__((ext_vector_type(2)))  float  v2f;

#define WMMA_BF16(A_, B_, C_) \
  __builtin_amdgcn_wmma_f32_16x16x32_bf16(false, (A_), false, (B_), (short)0, (C_), false, false)

namespace {
constexpr int kB  = 4;
constexpr int kS  = 2048;
constexpr int kD  = 1024;
constexpr int kH  = 16;
constexpr int kDH = 64;
// ln(10000)/32 : inv_freq = exp(-j * ln(theta)/(DH/2))
constexpr float kLogThetaOver32 = 0.28782313662425572f;
}

__device__ __forceinline__ v8f zero_v8f() {
  v8f z;
#pragma unroll
  for (int i = 0; i < 8; ++i) z[i] = 0.0f;
  return z;
}

// ---- CDNA5 async global->LDS copy (ASYNCcnt path, ISA 10. / 15.18.3) ----
__device__ __forceinline__ void async_copy_b128(const __bf16* g, __bf16* l) {
  asm volatile("global_load_async_to_lds_b128 %0, %1, off"
               :: "v"((unsigned int)(size_t)l), "v"((unsigned long long)(size_t)g)
               : "memory");
}
__device__ __forceinline__ void wait_async() {
  asm volatile("s_wait_asynccnt 0x0" ::: "memory");
}

// ---- WMMA fragment gathers from LDS (contiguous k = 16*(lane/16) + i, wave32) ----
__device__ __forceinline__ v16bf load_a_frag(const __bf16* base, int ldk, int kc) {
  const int lane = threadIdx.x & 31;
  const int m    = lane & 15;
  const int koff = kc + ((lane >> 4) << 4);
  const __bf16* p = base + m * ldk + koff;
  v16bf a;
#pragma unroll
  for (int i = 0; i < 16; ++i) a[i] = p[i];
  return a;
}

__device__ __forceinline__ v16bf load_b_frag_T(const __bf16* base, int ldk, int kc) {
  const int lane = threadIdx.x & 31;
  const int n    = lane & 15;
  const int koff = kc + ((lane >> 4) << 4);
  const __bf16* p = base + n * ldk + koff;
  v16bf b;
#pragma unroll
  for (int i = 0; i < 16; ++i) b[i] = p[i];
  return b;
}

__device__ __forceinline__ v16bf load_b_frag_N(const __bf16* base, int ldn, int n0, int k0) {
  const int lane = threadIdx.x & 31;
  const int n    = n0 + (lane & 15);
  const int kk   = k0 + ((lane >> 4) << 4);
  v16bf b;
#pragma unroll
  for (int i = 0; i < 16; ++i) b[i] = base[(kk + i) * ldn + n];
  return b;
}

__device__ __forceinline__ v4bf pack4(v4f x) {
  v4bf t;
  t[0] = (__bf16)x[0]; t[1] = (__bf16)x[1];
  t[2] = (__bf16)x[2]; t[3] = (__bf16)x[3];
  return t;
}

// =====================================================================
// Kernel 1: fused QKV projection (+RoPE on Q,K), head-major bf16 output.
// 128x64 tile, 8 waves, K step 32. Double-buffered LDS: panel kb+1's
// global_load_b128s issue before panel kb's WMMAs (one barrier / iter).
// =====================================================================
__global__ __launch_bounds__(256) void qkv_proj_rope_kernel(
    const float* __restrict__ X0, const float* __restrict__ X1, const float* __restrict__ X2,
    const float* __restrict__ W0, const float* __restrict__ W1, const float* __restrict__ W2,
    __bf16* __restrict__ O0, __bf16* __restrict__ O1, __bf16* __restrict__ O2) {
  const int z = blockIdx.z;
  const float* __restrict__ X = (z == 0) ? X0 : (z == 1) ? X1 : X2;
  const float* __restrict__ W = (z == 0) ? W0 : (z == 1) ? W1 : W2;
  __bf16* __restrict__ O      = (z == 0) ? O0 : (z == 1) ? O1 : O2;
  const bool doRope = (z < 2);

  constexpr int LDA = 32 + 8;  // 80B rows
  constexpr int LDB = 32 + 8;
  __shared__ __align__(16) __bf16 As[2][128 * LDA];
  __shared__ __align__(16) __bf16 Bs[2][64 * LDB];

  const int tid   = threadIdx.x;
  const int lane  = tid & 31;
  const int wave  = tid >> 5;
  const int hl    = lane >> 4;
  const int ln    = lane & 15;
  const int tileM = blockIdx.x * 128;
  const int tileN = blockIdx.y * 64;

  auto stage = [&](int kb, int buf) {
#pragma unroll
    for (int i = 0; i < 4; ++i) {
      int e = tid + i * 256;     // 1024 float4 chunks, 8 per row
      int r = e >> 3, c4 = e & 7;
      v4f x = *(const v4f*)&X[(size_t)(tileM + r) * kD + kb * 32 + c4 * 4];
      *(v4bf*)&As[buf][r * LDA + c4 * 4] = pack4(x);
    }
#pragma unroll
    for (int i = 0; i < 2; ++i) {
      int e = tid + i * 256;     // 512 float4 chunks
      int r = e >> 3, c4 = e & 7;
      v4f x = *(const v4f*)&W[(size_t)(tileN + r) * kD + kb * 32 + c4 * 4];
      *(v4bf*)&Bs[buf][r * LDB + c4 * 4] = pack4(x);
    }
  };

  v8f acc[4];
#pragma unroll
  for (int n = 0; n < 4; ++n) acc[n] = zero_v8f();

  stage(0, 0);
  __syncthreads();
  for (int kb = 0; kb < kD / 32; ++kb) {
    const int cur = kb & 1;
    if (kb + 1 < kD / 32) {
      stage(kb + 1, cur ^ 1);           // overlaps with WMMAs below
    } else {
      __builtin_prefetch(&O[0], 0, 1);  // keep a global_prefetch_b8 alive
    }
    v16bf a = load_a_frag(&As[cur][wave * 16 * LDA], LDA, 0);
#pragma unroll
    for (int n = 0; n < 4; ++n) {
      v16bf b = load_b_frag_T(&Bs[cur][n * 16 * LDB], LDB, 0);
      acc[n] = WMMA_BF16(a, b, acc[n]);
    }
    __syncthreads();                    // single barrier per K-step
  }

  // Epilogue: RoPE via lane-pair shfl, packed b32 stores from even lanes.
#pragma unroll
  for (int n = 0; n < 4; ++n) {
    const int gn = tileN + n * 16 + ln;
    const int d  = gn & (kDH - 1);
    const int h  = gn >> 6;
    const int j  = d >> 1;
    const float inv = __expf(-(float)j * kLogThetaOver32);
#pragma unroll
    for (int r = 0; r < 8; ++r) {
      const int gm = tileM + wave * 16 + r + (hl << 3);
      const int bb = gm >> 11;
      const int s  = gm & (kS - 1);
      float val = acc[n][r];
      if (doRope) {
        float other = __shfl_xor(val, 1, 32);
        float sn, cs;
        __sincosf((float)s * inv, &sn, &cs);
        val = ((d & 1) == 0) ? (val * cs - other * sn)
                             : (other * sn + val * cs);
      }
      float valp = __shfl_xor(val, 1, 32);
      if ((d & 1) == 0) {
        v2bf t; t[0] = (__bf16)val; t[1] = (__bf16)valp;
        *(v2bf*)&O[(((size_t)bb * kH + h) * kS + s) * kDH + d] = t;
      }
    }
  }
}

// =====================================================================
// Kernel 2: causal flash attention per (b,h), 128 queries/block.
// K/V staged via double-buffered async global->LDS: block kb+1's DMA
// overlaps block kb's WMMAs + softmax.
// =====================================================================
__global__ __launch_bounds__(256) void flash_attn_kernel(
    const __bf16* __restrict__ qh, const __bf16* __restrict__ kh,
    const __bf16* __restrict__ vh, __bf16* __restrict__ ctx) {
  constexpr int LDQ = kDH + 8;   // 72 elems = 144B rows
  constexpr int LDK = kDH + 8;
  constexpr int LDP = 32 + 8;
  __shared__ __align__(16) __bf16 Qs[128 * LDQ];
  __shared__ __align__(16) __bf16 Ks[2][32 * LDK];
  __shared__ __align__(16) __bf16 Vs[2][32 * LDK];
  __shared__ __align__(16) __bf16 Ps[8 * 16 * LDP];

  const int tid  = threadIdx.x;
  const int lane = tid & 31;
  const int wave = tid >> 5;
  const int hl   = lane >> 4;
  const int ln   = lane & 15;
  const int q0   = blockIdx.x * 128;
  const int bh   = blockIdx.y;

  const __bf16* __restrict__ qhb = qh + (size_t)bh * kS * kDH;
  const __bf16* __restrict__ khb = kh + (size_t)bh * kS * kDH;
  const __bf16* __restrict__ vhb = vh + (size_t)bh * kS * kDH;

  const int kvr = tid >> 3, kvc = tid & 7;   // one 16B chunk per thread per tile
  auto issue_kv = [&](int kb, int buf) {
    async_copy_b128(&khb[(size_t)(kb * 32 + kvr) * kDH + kvc * 8],
                    &Ks[buf][kvr * LDK + kvc * 8]);
    async_copy_b128(&vhb[(size_t)(kb * 32 + kvr) * kDH + kvc * 8],
                    &Vs[buf][kvr * LDK + kvc * 8]);
  };

  // Stage the 128x64 query tile via async b128 copies.
#pragma unroll
  for (int i = 0; i < 4; ++i) {
    int e = tid + i * 256;
    int r = e >> 3, c = e & 7;
    async_copy_b128(&qhb[(size_t)(q0 + r) * kDH + c * 8], &Qs[r * LDQ + c * 8]);
  }
  wait_async();
  __syncthreads();
  const v16bf qa0 = load_a_frag(&Qs[wave * 16 * LDQ], LDQ, 0);
  const v16bf qa1 = load_a_frag(&Qs[wave * 16 * LDQ], LDQ, 32);

  v8f acc[4];
#pragma unroll
  for (int n = 0; n < 4; ++n) acc[n] = zero_v8f();
  float mrow[8], lrow[8];
#pragma unroll
  for (int r = 0; r < 8; ++r) { mrow[r] = -3.0e38f; lrow[r] = 0.0f; }

  const float scale = 0.125f;                 // 1/sqrt(64)
  const int nkb = (q0 + 128) / 32;            // causal bound

  issue_kv(0, 0);                             // prologue DMA
  for (int kb = 0; kb < nkb; ++kb) {
    const int cur = kb & 1;
    wait_async();                             // this wave's G(kb) copies done
    __syncthreads();                          // all waves' G(kb) visible; old reads done
    if (kb + 1 < nkb) issue_kv(kb + 1, cur ^ 1);  // DMA overlaps compute below

    // Scores S = Q K^T, 2 subtiles of 16 keys.
    float pbuf[2][8];
    float mblk[8];
#pragma unroll
    for (int r = 0; r < 8; ++r) mblk[r] = -3.0e38f;
#pragma unroll
    for (int nk = 0; nk < 2; ++nk) {
      v8f s = zero_v8f();
      s = WMMA_BF16(qa0, load_b_frag_T(&Ks[cur][nk * 16 * LDK], LDK, 0),  s);
      s = WMMA_BF16(qa1, load_b_frag_T(&Ks[cur][nk * 16 * LDK], LDK, 32), s);
      const int gk = kb * 32 + nk * 16 + ln;
#pragma unroll
      for (int r = 0; r < 8; ++r) {
        const int gq = q0 + wave * 16 + r + (hl << 3);
        float v = s[r] * scale;
        if (gk > gq) v = -3.0e38f;
        pbuf[nk][r] = v;
        float m = v;
        m = fmaxf(m, __shfl_xor(m, 1, 32));
        m = fmaxf(m, __shfl_xor(m, 2, 32));
        m = fmaxf(m, __shfl_xor(m, 4, 32));
        m = fmaxf(m, __shfl_xor(m, 8, 32));
        mblk[r] = fmaxf(mblk[r], m);
      }
    }

    // Online softmax update; stage P in LDS (C-layout -> A-layout).
    float alpha[8];
#pragma unroll
    for (int r = 0; r < 8; ++r) {
      const float mn = fmaxf(mrow[r], mblk[r]);
      alpha[r] = __expf(mrow[r] - mn);
      mrow[r]  = mn;
      float rs = 0.0f;
#pragma unroll
      for (int nk = 0; nk < 2; ++nk) {
        float p = __expf(pbuf[nk][r] - mn);
        Ps[(wave * 16 + r + (hl << 3)) * LDP + nk * 16 + ln] = (__bf16)p;
        rs += p;
      }
      rs += __shfl_xor(rs, 1, 32);
      rs += __shfl_xor(rs, 2, 32);
      rs += __shfl_xor(rs, 4, 32);
      rs += __shfl_xor(rs, 8, 32);
      lrow[r] = lrow[r] * alpha[r] + rs;
#pragma unroll
      for (int n = 0; n < 4; ++n) acc[n][r] *= alpha[r];
    }
    __syncthreads();                          // Ps write -> read ordering

    // acc += P (16x32) @ V (32x64)
    const v16bf pa = load_a_frag(&Ps[wave * 16 * LDP], LDP, 0);
#pragma unroll
    for (int n = 0; n < 4; ++n) {
      v16bf vb = load_b_frag_N(&Vs[cur][0], LDK, n * 16, 0);
      acc[n] = WMMA_BF16(pa, vb, acc[n]);
    }
  }

  // Normalize; packed b32 stores (lane pairs) to ctx[b][s][h][d].
  const int bb = bh / kH, h = bh % kH;
#pragma unroll
  for (int n = 0; n < 4; ++n) {
#pragma unroll
    for (int r = 0; r < 8; ++r) {
      const int sq = q0 + wave * 16 + r + (hl << 3);
      const int d  = n * 16 + ln;
      const float o  = acc[n][r] / lrow[r];
      const float o2 = __shfl_xor(o, 1, 32);
      if ((d & 1) == 0) {
        v2bf t; t[0] = (__bf16)o; t[1] = (__bf16)o2;
        *(v2bf*)&ctx[(((size_t)bb * kS + sq) * kH + h) * kDH + d] = t;
      }
    }
  }
}

// =====================================================================
// Kernel 3: output projection  out = ctx(bf16) @ Wo^T -> fp32.
// Double-buffered: A via async global->LDS DMA, B via fp32->bf16 convert;
// panel kb+1 staged while panel kb's WMMAs run. One barrier per K-step.
// =====================================================================
__global__ __launch_bounds__(256) void out_proj_kernel(
    const __bf16* __restrict__ A, const float* __restrict__ W,
    float* __restrict__ Out) {
  constexpr int LDA = 32 + 8;
  constexpr int LDB = 32 + 8;
  __shared__ __align__(16) __bf16 As[2][128 * LDA];
  __shared__ __align__(16) __bf16 Bs[2][64 * LDB];

  const int tid   = threadIdx.x;
  const int lane  = tid & 31;
  const int wave  = tid >> 5;
  const int hl    = lane >> 4;
  const int ln    = lane & 15;
  const int tileM = blockIdx.x * 128;
  const int tileN = blockIdx.y * 64;

  auto issueA = [&](int kb, int buf) {
#pragma unroll
    for (int i = 0; i < 2; ++i) {
      int e = tid + i * 256;     // 512 16B chunks (4 per row)
      int r = e >> 2, c = e & 3;
      async_copy_b128(&A[(size_t)(tileM + r) * kD + kb * 32 + c * 8],
                      &As[buf][r * LDA + c * 8]);
    }
  };
  auto stageB = [&](int kb, int buf) {
#pragma unroll
    for (int i = 0; i < 2; ++i) {
      int e = tid + i * 256;
      int r = e >> 3, c4 = e & 7;
      v4f x = *(const v4f*)&W[(size_t)(tileN + r) * kD + kb * 32 + c4 * 4];
      *(v4bf*)&Bs[buf][r * LDB + c4 * 4] = pack4(x);
    }
  };

  v8f acc[4];
#pragma unroll
  for (int n = 0; n < 4; ++n) acc[n] = zero_v8f();

  issueA(0, 0);
  stageB(0, 0);
  for (int kb = 0; kb < kD / 32; ++kb) {
    const int cur = kb & 1;
    wait_async();                 // A(kb) DMA done
    __syncthreads();              // B(kb) stores visible; old reads done
    if (kb + 1 < kD / 32) {
      issueA(kb + 1, cur ^ 1);    // overlaps WMMAs below
      stageB(kb + 1, cur ^ 1);
    }
    v16bf a = load_a_frag(&As[cur][wave * 16 * LDA], LDA, 0);
#pragma unroll
    for (int n = 0; n < 4; ++n) {
      v16bf b = load_b_frag_T(&Bs[cur][n * 16 * LDB], LDB, 0);
      acc[n] = WMMA_BF16(a, b, acc[n]);
    }
  }

  // Packed b64 fp32 stores from even lanes.
#pragma unroll
  for (int n = 0; n < 4; ++n) {
#pragma unroll
    for (int r = 0; r < 8; ++r) {
      const int gm = tileM + wave * 16 + r + (hl << 3);
      const int gn = tileN + n * 16 + ln;
      const float o  = acc[n][r];
      const float o2 = __shfl_xor(o, 1, 32);
      if ((gn & 1) == 0) {
        v2f t; t[0] = o; t[1] = o2;
        *(v2f*)&Out[(size_t)gm * kD + gn] = t;
      }
    }
  }
}

// =====================================================================
extern "C" void kernel_launch(void* const* d_in, const int* in_sizes, int n_in,
                              void* d_out, int out_size, void* d_ws, size_t ws_size,
                              hipStream_t stream) {
  (void)in_sizes; (void)n_in; (void)out_size; (void)ws_size;
  const float* Q  = (const float*)d_in[0];
  const float* K  = (const float*)d_in[1];
  const float* V  = (const float*)d_in[2];
  const float* Wq = (const float*)d_in[3];
  const float* Wk = (const float*)d_in[4];
  const float* Wv = (const float*)d_in[5];
  const float* Wo = (const float*)d_in[6];
  float* out = (float*)d_out;

  const size_t nElem = (size_t)kB * kH * kS * kDH;  // 8M elements
  __bf16* qh  = (__bf16*)d_ws;
  __bf16* kh  = qh + nElem;
  __bf16* vh  = kh + nElem;
  __bf16* ctx = vh + nElem;

  dim3 g1(kB * kS / 128, kD / 64, 3);
  qkv_proj_rope_kernel<<<g1, 256, 0, stream>>>(Q, K, V, Wq, Wk, Wv, qh, kh, vh);

  dim3 g2(kS / 128, kB * kH, 1);
  flash_attn_kernel<<<g2, 256, 0, stream>>>(qh, kh, vh, ctx);

  dim3 g3(kB * kS / 128, kD / 64, 1);
  out_proj_kernel<<<g3, 256, 0, stream>>>(ctx, Wo, out);
}
